// ModernNet_54331336294747
// MI455X (gfx1250) — matrix operand order repaired and checked by reading, not compiled, not run on hardware
//
#include <hip/hip_runtime.h>
#include <hip/hip_bf16.h>

typedef __attribute__((ext_vector_type(16))) _Float16 v16h;
typedef __attribute__((ext_vector_type(8)))  _Float16 v8h;
typedef __attribute__((ext_vector_type(4)))  _Float16 v4h;
typedef __attribute__((ext_vector_type(8)))  float    v8f;

#define WMMA_F16(a, b, c) \
  __builtin_amdgcn_wmma_f32_16x16x32_f16(false, (a), false, (b), (short)0, (c), false, false)

// ---- fragment gathers (CDNA5 wave32 16x16x32 f16, contiguous k = 16h + s in both A and B) ----
// A 16x32: lane = m + 16h; VGPR j holds K = 16h + 2j, 16h + 2j + 1
__device__ inline v16h gather_a(const _Float16* p, int sl, int sk, int lane) {
  int mm = lane & 15, h = lane >> 4;
  v16h f;
#pragma unroll
  for (int j = 0; j < 8; ++j) {
    int k = 16 * h + 2 * j;
    f[2 * j]     = p[mm * sl + k * sk];
    f[2 * j + 1] = p[mm * sl + (k + 1) * sk];
  }
  return f;
}
// B 32x16: lane = n + 16h; VGPR j holds K = 16h + 2j, 16h + 2j + 1
// source laid out n-major (sBt[n][k]) -> 16 contiguous halfs per lane
__device__ inline v16h gather_b(const _Float16* p, int sl, int sk, int lane) {
  int nn = lane & 15, h = lane >> 4;
  v16h f;
#pragma unroll
  for (int j = 0; j < 8; ++j) {
    int k = 16 * h + 2 * j;
    f[2 * j]     = p[nn * sl + k * sk];
    f[2 * j + 1] = p[nn * sl + (k + 1) * sk];
  }
  return f;
}

__device__ inline v4h cvt4(float4 v) {
  v4h r;
  r[0] = (_Float16)v.x; r[1] = (_Float16)v.y;
  r[2] = (_Float16)v.z; r[3] = (_Float16)v.w;
  return r;
}

__device__ inline void atomicMaxF32(float* addr, float v) {
  if (v >= 0.f) atomicMax((int*)addr, __float_as_int(v));
  else          atomicMin((unsigned int*)addr, (unsigned int)__float_as_int(v));
}

// ---- init: deg = 0, m = -inf ----
__global__ void k_init(float* deg, float* m, int ndeg, int nm) {
  int i = blockIdx.x * blockDim.x + threadIdx.x;
  if (i < ndeg) deg[i] = 0.f;
  if (i < nm)   m[i] = -__builtin_inff();
}

// ---- GEMM1: X[a,c] = sum_l seq[a,l]*adj[c,l]; deg partials from registers ----
__global__ __launch_bounds__(256) void k_gemm1(const float* __restrict__ seq,
                                               const float* __restrict__ adj,
                                               _Float16* __restrict__ Xr,
                                               float* __restrict__ deg) {
  const int b  = blockIdx.y;
  const int c0 = blockIdx.x * 128;
  const int tid = threadIdx.x, wave = tid >> 5, lane = tid & 31;

  __shared__ _Float16 sA[32][512];   // seq padded to 32 rows
  __shared__ _Float16 sB[128][72];   // adj chunk (n-major): 128 c x 64 l
  __shared__ float    sdeg[512];     // per-WG degree partials

  for (int i = tid; i < 512; i += 256) sdeg[i] = 0.f;

  const float* seqb = seq + (size_t)b * 20 * 512;
#pragma unroll
  for (int j = 0; j < 16; ++j) {
    int i = tid + j * 256, a = i >> 7, q = i & 127;
    float4 v = make_float4(0.f, 0.f, 0.f, 0.f);
    if (a < 20) v = *(const float4*)(seqb + a * 512 + q * 4);
    *(v4h*)&sA[a][q * 4] = cvt4(v);
  }

  const float* adjb = adj + (size_t)b * 2048 * 512;
  const int mt = wave & 1;        // M-tile (a 0..15 / 16..31)
  const int np = wave >> 1;       // owns N-tiles np*2, np*2+1
  v8f z = {};
  v8f acc0 = z, acc1 = z;

  const int rr = tid >> 4;        // base row (adds 16 per batch step)
  const int qq = tid & 15;        // fixed l-quad per thread

  {  // prefetch first chunk
    int r = tid >> 1, pt = tid & 1;
    __builtin_prefetch(adjb + (size_t)(c0 + r) * 512 + pt * 32, 0, 0);
  }

  for (int ch = 0; ch < 8; ++ch) {             // K = 512 in chunks of 64
    __syncthreads();
    // batched loads: 8 x float4 per thread, then convert+store
    float4 tmp[8];
#pragma unroll
    for (int j = 0; j < 8; ++j)
      tmp[j] = *(const float4*)(adjb + (size_t)(c0 + rr + j * 16) * 512 + ch * 64 + qq * 4);
    // degree partials over this thread's 8 rows (f32, matches reference)
    {
      float s0 = 0.f, s1 = 0.f, s2 = 0.f, s3 = 0.f;
#pragma unroll
      for (int j = 0; j < 8; ++j) { s0 += tmp[j].x; s1 += tmp[j].y; s2 += tmp[j].z; s3 += tmp[j].w; }
      atomicAdd(&sdeg[ch * 64 + qq * 4 + 0], s0);
      atomicAdd(&sdeg[ch * 64 + qq * 4 + 1], s1);
      atomicAdd(&sdeg[ch * 64 + qq * 4 + 2], s2);
      atomicAdd(&sdeg[ch * 64 + qq * 4 + 3], s3);
    }
#pragma unroll
    for (int j = 0; j < 8; ++j)
      *(v4h*)&sB[rr + j * 16][qq * 4] = cvt4(tmp[j]);
    if (ch < 7) {                               // prefetch next chunk
      int r = tid >> 1, pt = tid & 1;
      __builtin_prefetch(adjb + (size_t)(c0 + r) * 512 + (ch + 1) * 64 + pt * 32, 0, 0);
    }
    __syncthreads();
#pragma unroll
    for (int s = 0; s < 2; ++s) {
      v16h af  = gather_a(&sA[mt * 16][ch * 64 + s * 32], 512, 1, lane);
      v16h bf0 = gather_b(&sB[(np * 2 + 0) * 16][s * 32], 72, 1, lane); // B = adj^T
      v16h bf1 = gather_b(&sB[(np * 2 + 1) * 16][s * 32], 72, 1, lane);
      acc0 = WMMA_F16(af, bf0, acc0);
      acc1 = WMMA_F16(af, bf1, acc1);
    }
  }
  __syncthreads();
  for (int l = tid; l < 512; l += 256) atomicAdd(&deg[b * 512 + l], sdeg[l]);

  // store with conv reshape: X[a][c] -> Xr[a*2 + (c&1)][c>>1]
  int n = lane & 15, h = lane >> 4;
  _Float16* Xrb = Xr + (size_t)b * 40 * 1024;
#pragma unroll
  for (int t = 0; t < 2; ++t) {
    v8f ac = t ? acc1 : acc0;
    int c = c0 + (np * 2 + t) * 16 + n;
    int cc = c >> 1, par = c & 1;
#pragma unroll
    for (int r = 0; r < 8; ++r) {
      int a = mt * 16 + r + 8 * h;
      if (a < 20) Xrb[(a * 2 + par) * 1024 + cc] = (_Float16)ac[r];
    }
  }
}

// ---- GEMM2: Y(64x1024) = Wc(64x40,pad64) @ Xr(40x1024,pad64) ----
__global__ __launch_bounds__(256) void k_gemm2(const float* __restrict__ wconv,
                                               const _Float16* __restrict__ Xr,
                                               _Float16* __restrict__ Y) {
  const int b = blockIdx.y;
  const int cc0 = blockIdx.x * 128;
  const int tid = threadIdx.x, wave = tid >> 5, lane = tid & 31;
  __shared__ _Float16 sA[64][72];    // w_conv padded 40->64
  __shared__ _Float16 sBt[128][72];  // Xr chunk transposed: [n][k], cols >=40 zero

  for (int i = tid; i < 64 * 64; i += 256) {
    int f = i >> 6, j = i & 63;
    sA[f][j] = (j < 40) ? (_Float16)wconv[f * 40 + j] : (_Float16)0.f;
  }
  const _Float16* Xrb = Xr + (size_t)b * 40 * 1024;
  {
    const int qq = tid & 15;       // fixed n-octet per thread
    v8h tmp[4];
#pragma unroll
    for (int j = 0; j < 4; ++j) {
      int k = (tid >> 4) + j * 16;
      v8h zz = {};
      tmp[j] = (k < 40) ? *(const v8h*)(Xrb + k * 1024 + cc0 + qq * 8) : zz;
    }
#pragma unroll
    for (int j = 0; j < 4; ++j) {
      int k = (tid >> 4) + j * 16;
#pragma unroll
      for (int e = 0; e < 8; ++e) sBt[qq * 8 + e][k] = tmp[j][e];
    }
  }
  __syncthreads();

  const int mt = wave & 3, nh = wave >> 2;
  v8f z = {};
  v8f acc[4] = {z, z, z, z};
#pragma unroll
  for (int s = 0; s < 2; ++s) {
    v16h af = gather_a(&sA[mt * 16][s * 32], 72, 1, lane);
#pragma unroll
    for (int t = 0; t < 4; ++t) {
      v16h bf = gather_b(&sBt[(nh * 4 + t) * 16][s * 32], 72, 1, lane);
      acc[t] = WMMA_F16(af, bf, acc[t]);
    }
  }
  int n = lane & 15, h = lane >> 4;
  _Float16* Yb = Y + (size_t)b * 64 * 1024;
#pragma unroll
  for (int t = 0; t < 4; ++t) {
    int cc = cc0 + (nh * 4 + t) * 16 + n;
#pragma unroll
    for (int r = 0; r < 8; ++r) {
      int f = mt * 16 + r + 8 * h;
      Yb[f * 1024 + cc] = (_Float16)acc[t][r];
    }
  }
}

// ---- GEMM3: Z(64x128 tile) = Y(64x1024) @ adj_even(1024x128); /deg; max -> m ----
__global__ __launch_bounds__(256) void k_gemm3(const _Float16* __restrict__ Y,
                                               const float* __restrict__ adj,
                                               const float* __restrict__ deg,
                                               float* __restrict__ m) {
  const int b = blockIdx.y;
  const int l0 = blockIdx.x * 128;
  const int tid = threadIdx.x, wave = tid >> 5, lane = tid & 31;
  __shared__ _Float16 sA[64][72];    // Y chunk [f][k]
  __shared__ _Float16 sBt[128][72];  // adj_even chunk transposed: [l][k]
  __shared__ float    zbuf[64][132];

  const _Float16* Yb = Y + (size_t)b * 64 * 1024;
  const float* adjb = adj + (size_t)b * 2048 * 512;
  const int mt = wave & 3, nh = wave >> 2;
  v8f z = {};
  v8f acc[4] = {z, z, z, z};

  {  // prefetch first chunk of adj_even
    int kk = tid >> 2, pt = tid & 3;
    __builtin_prefetch(adjb + (size_t)(2 * kk) * 512 + l0 + pt * 32, 0, 0);
  }

  const int qq = tid & 31;         // fixed l-quad per thread (B staging)

  for (int ch = 0; ch < 16; ++ch) {          // K = 1024 in chunks of 64
    __syncthreads();
    // batched loads: Y chunk (2 x 16B) + adj_even chunk (8 x float4)
    v8h ta0 = *(const v8h*)(Yb + (tid >> 3) * 1024 + ch * 64 + (tid & 7) * 8);
    v8h ta1 = *(const v8h*)(Yb + ((tid >> 3) + 32) * 1024 + ch * 64 + (tid & 7) * 8);
    float4 tb[8];
#pragma unroll
    for (int j = 0; j < 8; ++j) {
      int kk = (tid >> 5) + j * 8;
      tb[j] = *(const float4*)(adjb + (size_t)(2 * (ch * 64 + kk)) * 512 + l0 + qq * 4);
    }
    *(v8h*)&sA[tid >> 3][(tid & 7) * 8] = ta0;
    *(v8h*)&sA[(tid >> 3) + 32][(tid & 7) * 8] = ta1;
#pragma unroll
    for (int j = 0; j < 8; ++j) {
      int kk = (tid >> 5) + j * 8;
      v4h h4 = cvt4(tb[j]);
      sBt[qq * 4 + 0][kk] = h4[0];
      sBt[qq * 4 + 1][kk] = h4[1];
      sBt[qq * 4 + 2][kk] = h4[2];
      sBt[qq * 4 + 3][kk] = h4[3];
    }
    if (ch < 15) {                            // prefetch next chunk
      int kk = tid >> 2, pt = tid & 3;
      __builtin_prefetch(adjb + (size_t)(2 * ((ch + 1) * 64 + kk)) * 512 + l0 + pt * 32, 0, 0);
    }
    __syncthreads();
#pragma unroll
    for (int s = 0; s < 2; ++s) {
      v16h af = gather_a(&sA[mt * 16][s * 32], 72, 1, lane);
#pragma unroll
      for (int t = 0; t < 4; ++t) {
        v16h bf = gather_b(&sBt[(nh * 4 + t) * 16][s * 32], 72, 1, lane);
        acc[t] = WMMA_F16(af, bf, acc[t]);
      }
    }
  }
  __syncthreads();
  int n = lane & 15, h = lane >> 4;
#pragma unroll
  for (int t = 0; t < 4; ++t) {
    int col = (nh * 4 + t) * 16 + n;
    float d = deg[b * 512 + l0 + col];
#pragma unroll
    for (int r = 0; r < 8; ++r) {
      int f = mt * 16 + r + 8 * h;
      zbuf[f][col] = acc[t][r] / d;
    }
  }
  __syncthreads();
  if (tid < 64) {
    float mx = -__builtin_inff();
    for (int col = 0; col < 128; ++col) mx = fmaxf(mx, zbuf[tid][col]);
    atomicMaxF32(&m[b * 64 + tid], mx);
  }
}

// ---- final: out[b] = sigmoid(sum_f m[b,f] * wcomb[f]) ----
__global__ void k_final(const float* __restrict__ m, const float* __restrict__ wcomb,
                        float* __restrict__ out) {
  int b = blockIdx.x, f = threadIdx.x;
  __shared__ float red[64];
  red[f] = m[b * 64 + f] * wcomb[f];
  __syncthreads();
  for (int s = 32; s > 0; s >>= 1) {
    if (f < s) red[f] += red[f + s];
    __syncthreads();
  }
  if (f == 0) out[b] = 1.f / (1.f + expf(-red[0]));
}

extern "C" void kernel_launch(void* const* d_in, const int* in_sizes, int n_in,
                              void* d_out, int out_size, void* d_ws, size_t ws_size,
                              hipStream_t stream) {
  const float* seq   = (const float*)d_in[0];   // [128,20,512]
  const float* adj   = (const float*)d_in[1];   // [128,2048,512]
  const float* wconv = (const float*)d_in[2];   // [64,20,2]
  const float* wcomb = (const float*)d_in[3];   // [1,64]
  float* out = (float*)d_out;                   // [128]

  char* ws = (char*)d_ws;
  const size_t XR_BYTES = (size_t)128 * 40 * 1024 * 2;   // 10 MB f16
  const size_t Y_BYTES  = (size_t)128 * 64 * 1024 * 2;   // 16 MB f16
  const size_t DEG_BYTES = (size_t)128 * 512 * 4;
  _Float16* Xr  = (_Float16*)ws;
  _Float16* Yw  = (_Float16*)(ws + XR_BYTES);
  float*    deg = (float*)(ws + XR_BYTES + Y_BYTES);
  float*    mbuf = (float*)(ws + XR_BYTES + Y_BYTES + DEG_BYTES);

  k_init<<<256, 256, 0, stream>>>(deg, mbuf, 128 * 512, 128 * 64);
  dim3 g1(16, 128); k_gemm1<<<g1, 256, 0, stream>>>(seq, adj, Xr, deg);
  dim3 g2(8, 128);  k_gemm2<<<g2, 256, 0, stream>>>(wconv, Xr, Yw);
  dim3 g3(4, 128);  k_gemm3<<<g3, 256, 0, stream>>>(Yw, adj, deg, mbuf);
  k_final<<<128, 64, 0, stream>>>(mbuf, wcomb, out);
}
